// DecoupledBottleneckAttention_84782654423115
// MI455X (gfx1250) — matrix operand order repaired and checked by reading, not compiled, not run on hardware
//
#include <hip/hip_runtime.h>

// ---------------------------------------------------------------------------
// DecoupledBottleneckAttention forward for gfx1250 (MI455X, wave32, WMMA).
// B=1, T=2048, D=1024, H=16, sem_hd=geo_hd=32, v_hd=64, Kctx = T+1 (null key).
// All matmuls via v_wmma_f32_16x16x32_bf16 (bf16 in, fp32 accumulate).
// Softmax row reductions via DPP ROW_XMASK (VALU butterflies, no LDS).
// Workspace (bf16): q_sem,k_sem,q_geo,k_geo (2MB each) + v (4MB) + ctx (4MB) = 16MB.
// ---------------------------------------------------------------------------

typedef __attribute__((ext_vector_type(16))) __bf16 v16bf;
typedef __attribute__((ext_vector_type(8)))  float  v8f;

#define TSEQ 2048
#define DMODEL 1024
#define NHEAD 16
#define HD32 32
#define VHD 64
#define NEGVAL -1000000000.0f
#define SCALE 0.17677669529663687f  /* 1/sqrt(32) */

__device__ __forceinline__ float bf2f(unsigned short u) {
    unsigned int x = ((unsigned int)u) << 16;
    float f; __builtin_memcpy(&f, &x, 4); return f;
}
__device__ __forceinline__ unsigned short f2bf(float f) {
    unsigned int x; __builtin_memcpy(&x, &f, 4);
    unsigned int r = x + 0x7FFFu + ((x >> 16) & 1u);
    return (unsigned short)(r >> 16);
}

union FragBF { v16bf v; unsigned short u[16]; };

// DPP ROW_XMASK butterfly within each 16-lane row: lane reads lane^MASK. Pure VALU.
template <int MASK>
__device__ __forceinline__ float dpp_xor(float v) {
    int x = __builtin_bit_cast(int, v);
    int y = __builtin_amdgcn_update_dpp(0, x, 0x160 + MASK, 0xf, 0xf, true);
    return __builtin_bit_cast(float, y);
}
template <int MASK>
__device__ __forceinline__ void red_max8(float* v) {
#pragma unroll
    for (int r = 0; r < 8; ++r) v[r] = fmaxf(v[r], dpp_xor<MASK>(v[r]));
}
template <int MASK>
__device__ __forceinline__ void red_add8(float* v) {
#pragma unroll
    for (int r = 0; r < 8; ++r) v[r] += dpp_xor<MASK>(v[r]);
}

// A-fragment (16x32 bf16, row-major source, leading dim ld halves):
// lane<16: row=lane, K=0..15; lane>=16: row=lane-16, K=16..31  (sequential K)
__device__ __forceinline__ v16bf load_a_bf16(const unsigned short* base, int ld, int lane) {
    FragBF r;
    int row = lane & 15;
    int k0  = (lane >> 4) * 16;
    const unsigned short* p = base + row * ld + k0;
#pragma unroll
    for (int i = 0; i < 8; ++i) { r.u[i] = p[i]; r.u[8 + i] = p[8 + i]; }
    return r.v;
}

// B-fragment (32x16 bf16) where B = Ksrc^T: B[k][n] = Ksrc[n][k], Ksrc row-major.
// lane<16: col=lane, K=0..15; lane>=16: col=lane-16, K=16..31  (sequential K)
__device__ __forceinline__ v16bf load_bT_bf16(const unsigned short* rowptr, int k0) {
    FragBF r;
    const unsigned short* p = rowptr + k0;
#pragma unroll
    for (int i = 0; i < 16; ++i) r.u[i] = p[i];
    return r.v;
}

__device__ __forceinline__ v8f wmma_bf16(v16bf a, v16bf b, v8f c) {
    return __builtin_amdgcn_wmma_f32_16x16x32_bf16(false, a, false, b, (short)0, c, false, false);
}

// ---------------------------------------------------------------------------
// Generic GEMM:  out[M,N] = A[M,K] @ W[K,N]   (W always fp32; A fp32 or bf16)
// BM=64 BN=64 BK=32, 128 threads (4 waves), each wave -> 16x64 of C.
// ---------------------------------------------------------------------------
template <bool A_BF16, bool OUT_F32>
__global__ __launch_bounds__(128) void gemm_wmma_kernel(const void* __restrict__ Ap,
                                                        const float* __restrict__ W,
                                                        void* __restrict__ outp,
                                                        int M, int N, int K) {
    __shared__ unsigned short lA[64 * 32];   // [m][k] bf16
    __shared__ unsigned short lB[64 * 32];   // [n][k] bf16 (transposed W tile)

    const int bm = blockIdx.y * 64;
    const int bn = blockIdx.x * 64;
    const int tid = threadIdx.x;
    const int lane = tid & 31;
    const int wave = tid >> 5;
    const int l16 = lane & 15;
    const int half = lane >> 4;

    const float*          Af  = (const float*)Ap;
    const unsigned short* Abf = (const unsigned short*)Ap;

    v8f acc[4];
#pragma unroll
    for (int j = 0; j < 4; ++j) acc[j] = (v8f){0.f, 0.f, 0.f, 0.f, 0.f, 0.f, 0.f, 0.f};

    for (int k0 = 0; k0 < K; k0 += 32) {
        // stage A tile: vectorized (float4 / 4x bf16) loads, packed 8B LDS stores
#pragma unroll
        for (int idx = 0; idx < 4; ++idx) {
            int i = tid * 4 + idx * 512;               // 2048 halves total
            int m = i >> 5, kk = i & 31;               // kk multiple of 4
            unsigned int lo, hi;
            if (A_BF16) {
                const unsigned short* p = Abf + (bm + m) * K + k0 + kk;
                uint2 raw = *(const uint2*)p;
                lo = raw.x; hi = raw.y;
            } else {
                const float* p = Af + (bm + m) * K + k0 + kk;
                float4 f = *(const float4*)p;
                lo = (unsigned int)f2bf(f.x) | ((unsigned int)f2bf(f.y) << 16);
                hi = (unsigned int)f2bf(f.z) | ((unsigned int)f2bf(f.w) << 16);
            }
            *(uint2*)&lA[i] = make_uint2(lo, hi);
        }
        // stage W tile transposed: coalesced float4 loads, scattered u16 LDS stores
#pragma unroll
        for (int idx = 0; idx < 4; ++idx) {
            int i = tid * 4 + idx * 512;
            int kk = i >> 6, n = i & 63;               // n multiple of 4
            float4 f = *(const float4*)&W[(k0 + kk) * N + bn + n];
            lB[(n + 0) * 32 + kk] = f2bf(f.x);
            lB[(n + 1) * 32 + kk] = f2bf(f.y);
            lB[(n + 2) * 32 + kk] = f2bf(f.z);
            lB[(n + 3) * 32 + kk] = f2bf(f.w);
        }
        __syncthreads();

        v16bf afrag = load_a_bf16(lA + (wave * 16) * 32, 32, lane);
#pragma unroll
        for (int j = 0; j < 4; ++j) {
            v16bf bfrag = load_bT_bf16(lB + (16 * j + l16) * 32, half * 16);
            acc[j] = wmma_bf16(afrag, bfrag, acc[j]);
        }
        __syncthreads();
    }

    // epilogue: C element (r,lane): row = r + half*8, col = l16 (within 16x16 tile j)
#pragma unroll
    for (int j = 0; j < 4; ++j) {
#pragma unroll
        for (int r = 0; r < 8; ++r) {
            int row = bm + wave * 16 + r + half * 8;
            int col = bn + 16 * j + l16;
            float vv = acc[j][r];
            if (OUT_F32) ((float*)outp)[row * N + col] = vv;
            else         ((unsigned short*)outp)[row * N + col] = f2bf(vv);
        }
    }
}

// ---------------------------------------------------------------------------
// RoPE in-place on bf16 buffer [T][H*32] (geo_hd=32 -> 16 rotation pairs).
// ---------------------------------------------------------------------------
__global__ __launch_bounds__(256) void rope_kernel(unsigned short* __restrict__ buf) {
    int idx = blockIdx.x * blockDim.x + threadIdx.x;   // T*H*16 total
    if (idx >= TSEQ * NHEAD * 16) return;
    int t = idx / (NHEAD * 16);
    int rem = idx - t * (NHEAD * 16);
    int h = rem >> 4;
    int d = rem & 15;
    float inv_freq = __expf(-(float)d * 0.5756462732485114f);  // 10000^(-d/16)
    float ang = (float)t * inv_freq;
    float s, c;
    __sincosf(ang, &s, &c);
    unsigned short* p = buf + t * (NHEAD * HD32) + h * HD32;
    float x1 = bf2f(p[d]);
    float x2 = bf2f(p[d + 16]);
    p[d]      = f2bf(x1 * c - x2 * s);
    p[d + 16] = f2bf(x2 * c + x1 * s);
}

// ---------------------------------------------------------------------------
// Flash attention: one wave per (16-query tile, head).
// scores = (Qs Ks^T + Qg Kg^T) * SCALE, null key prepended, causal mask,
// online softmax (DPP row butterflies), O = P V, ctx written bf16 [T][H*64].
// ---------------------------------------------------------------------------
__global__ __launch_bounds__(32) void attn_kernel(const unsigned short* __restrict__ qs,
                                                  const unsigned short* __restrict__ ks,
                                                  const unsigned short* __restrict__ qg,
                                                  const unsigned short* __restrict__ kg,
                                                  const unsigned short* __restrict__ vv,
                                                  const float* __restrict__ ksn,
                                                  const float* __restrict__ kgn,
                                                  const float* __restrict__ vn,
                                                  unsigned short* __restrict__ ctx) {
    __shared__ unsigned short lp[16 * 32];   // P tile staged for C-layout -> A-layout

    const int h    = blockIdx.y;
    const int qb   = blockIdx.x * 16;
    const int lane = threadIdx.x;
    const int l16  = lane & 15;
    const int half = lane >> 4;

    // Q fragments (persist in registers)
    v16bf aqs = load_a_bf16(qs + qb * 512 + h * HD32, 512, lane);
    v16bf aqg = load_a_bf16(qg + qb * 512 + h * HD32, 512, lane);

    // null score per query row (row = lane&15, duplicated across halves)
    float sn = 0.f;
    {
        const unsigned short* qsrow = qs + (qb + l16) * 512 + h * HD32;
        const unsigned short* qgrow = qg + (qb + l16) * 512 + h * HD32;
#pragma unroll
        for (int d = 0; d < HD32; ++d) {
            sn += bf2f(qsrow[d]) * ksn[h * HD32 + d];
            sn += bf2f(qgrow[d]) * kgn[h * HD32 + d];
        }
        sn *= SCALE;
    }

    // online-softmax state: rows r -> global row (half*8 + r); replicated per 16-lane half
    float m[8], l[8];
    v8f acc[4];
#pragma unroll
    for (int r = 0; r < 8; ++r) {
        m[r] = __shfl(sn, half * 8 + r, 32);
        l[r] = 1.0f;                              // exp(sn - sn) for the null key
    }
#pragma unroll
    for (int j = 0; j < 4; ++j) {
        float v0 = vn[h * VHD + 16 * j + l16];    // O init = 1 * v_null (broadcast over rows)
#pragma unroll
        for (int r = 0; r < 8; ++r) acc[j][r] = v0;
    }

    const v8f zero = (v8f){0.f, 0.f, 0.f, 0.f, 0.f, 0.f, 0.f, 0.f};

    for (int kb = 0; kb < qb + 16; kb += 32) {
        // prefetch next key block (K and V rows) into cache
        if (kb + 32 < qb + 16) {
            __builtin_prefetch(ks + (kb + 32 + lane) * 512 + h * HD32, 0, 1);
            __builtin_prefetch(kg + (kb + 32 + lane) * 512 + h * HD32, 0, 1);
            __builtin_prefetch(vv + (kb + 32 + lane) * DMODEL + h * VHD, 0, 1);
        }

        // --- scores: two 16x16 tiles over 32 keys, sem + geo chained ---
        v16bf bks0 = load_bT_bf16(ks + (kb +      l16) * 512 + h * HD32, half * 16);
        v16bf bkg0 = load_bT_bf16(kg + (kb +      l16) * 512 + h * HD32, half * 16);
        v16bf bks1 = load_bT_bf16(ks + (kb + 16 + l16) * 512 + h * HD32, half * 16);
        v16bf bkg1 = load_bT_bf16(kg + (kb + 16 + l16) * 512 + h * HD32, half * 16);

        v8f s0 = wmma_bf16(aqs, bks0, zero);
        s0     = wmma_bf16(aqg, bkg0, s0);
        v8f s1 = wmma_bf16(aqs, bks1, zero);
        s1     = wmma_bf16(aqg, bkg1, s1);

        const bool need_mask = (kb + 31 > qb);
        float a0[8], a1[8], rm[8];
#pragma unroll
        for (int r = 0; r < 8; ++r) {
            a0[r] = s0[r] * SCALE;
            a1[r] = s1[r] * SCALE;
            if (need_mask) {
                int q = qb + half * 8 + r;
                if (kb + l16      > q) a0[r] = NEGVAL;
                if (kb + 16 + l16 > q) a1[r] = NEGVAL;
            }
            rm[r] = fmaxf(a0[r], a1[r]);
        }
        // row max across 16 lanes: DPP butterflies, stage-outer so rows pipeline
        red_max8<1>(rm); red_max8<2>(rm); red_max8<4>(rm); red_max8<8>(rm);

        float alpha[8], p0[8], p1[8], rs[8];
#pragma unroll
        for (int r = 0; r < 8; ++r) {
            float mn = fmaxf(m[r], rm[r]);
            alpha[r] = __expf(m[r] - mn);
            m[r] = mn;
            p0[r] = __expf(a0[r] - mn);
            p1[r] = __expf(a1[r] - mn);
            rs[r] = p0[r] + p1[r];
        }
        red_add8<1>(rs); red_add8<2>(rs); red_add8<4>(rs); red_add8<8>(rs);
#pragma unroll
        for (int r = 0; r < 8; ++r) l[r] = l[r] * alpha[r] + rs[r];

        // rescale output accumulators
#pragma unroll
        for (int j = 0; j < 4; ++j)
#pragma unroll
            for (int r = 0; r < 8; ++r) acc[j][r] *= alpha[r];

        // --- C-layout P -> LDS -> A-layout fragment ---
#pragma unroll
        for (int r = 0; r < 8; ++r) {
            lp[(half * 8 + r) * 32 + l16]      = f2bf(p0[r]);
            lp[(half * 8 + r) * 32 + 16 + l16] = f2bf(p1[r]);
        }
        __syncthreads();
        v16bf ap = load_a_bf16(lp, 32, lane);
        __syncthreads();

        // --- O += P V : B[k][n] = V[kb+k][h*64 + 16j + n], K sequential per lane ---
#pragma unroll
        for (int j = 0; j < 4; ++j) {
            FragBF bv;
#pragma unroll
            for (int i = 0; i < 16; ++i)
                bv.u[i] = vv[(kb + half * 16 + i) * DMODEL + h * VHD + 16 * j + l16];
            acc[j] = wmma_bf16(ap, bv.v, acc[j]);
        }
    }

    // normalize and store ctx (bf16, [T][H*64])
#pragma unroll
    for (int j = 0; j < 4; ++j) {
#pragma unroll
        for (int r = 0; r < 8; ++r) {
            int row = qb + half * 8 + r;
            ctx[row * DMODEL + h * VHD + 16 * j + l16] = f2bf(acc[j][r] / l[r]);
        }
    }
}

// ---------------------------------------------------------------------------
extern "C" void kernel_launch(void* const* d_in, const int* in_sizes, int n_in,
                              void* d_out, int out_size, void* d_ws, size_t ws_size,
                              hipStream_t stream) {
    const float* x      = (const float*)d_in[0];
    const float* wq_sem = (const float*)d_in[1];
    const float* wk_sem = (const float*)d_in[2];
    const float* wq_geo = (const float*)d_in[3];
    const float* wk_geo = (const float*)d_in[4];
    const float* wv     = (const float*)d_in[5];
    const float* wo     = (const float*)d_in[6];
    const float* ksn    = (const float*)d_in[7];
    const float* kgn    = (const float*)d_in[8];
    const float* vnul   = (const float*)d_in[9];
    float* out = (float*)d_out;

    // workspace carve-up (bf16 elements); total 8M elems = 16 MB
    unsigned short* ws = (unsigned short*)d_ws;
    unsigned short* qsm = ws;                       // 2048*512
    unsigned short* kse = qsm + TSEQ * 512;
    unsigned short* qg  = kse + TSEQ * 512;
    unsigned short* kge = qg  + TSEQ * 512;
    unsigned short* vbf = kge + TSEQ * 512;         // 2048*1024
    unsigned short* ctx = vbf + TSEQ * DMODEL;      // 2048*1024

    // 1) projections: fp32 A, bf16 out
    dim3 blk(128);
    gemm_wmma_kernel<false, false><<<dim3(512 / 64, TSEQ / 64), blk, 0, stream>>>(x, wq_sem, qsm, TSEQ, 512, DMODEL);
    gemm_wmma_kernel<false, false><<<dim3(512 / 64, TSEQ / 64), blk, 0, stream>>>(x, wk_sem, kse, TSEQ, 512, DMODEL);
    gemm_wmma_kernel<false, false><<<dim3(512 / 64, TSEQ / 64), blk, 0, stream>>>(x, wq_geo, qg,  TSEQ, 512, DMODEL);
    gemm_wmma_kernel<false, false><<<dim3(512 / 64, TSEQ / 64), blk, 0, stream>>>(x, wk_geo, kge, TSEQ, 512, DMODEL);
    gemm_wmma_kernel<false, false><<<dim3(DMODEL / 64, TSEQ / 64), blk, 0, stream>>>(x, wv, vbf, TSEQ, DMODEL, DMODEL);

    // 2) RoPE on geo q/k (in-place, bf16)
    int rope_n = TSEQ * NHEAD * 16;
    rope_kernel<<<(rope_n + 255) / 256, 256, 0, stream>>>(qg);
    rope_kernel<<<(rope_n + 255) / 256, 256, 0, stream>>>(kge);

    // 3) flash attention: grid (T/16, H), wave32 blocks
    attn_kernel<<<dim3(TSEQ / 16, NHEAD), 32, 0, stream>>>(qsm, kse, qg, kge, vbf,
                                                           ksn, kgn, vnul, ctx);

    // 4) output projection: bf16 A, fp32 out
    gemm_wmma_kernel<true, true><<<dim3(DMODEL / 64, TSEQ / 64), blk, 0, stream>>>(ctx, wo, out, TSEQ, DMODEL, DMODEL);
}
